// EMGTLC_75771813036619
// MI455X (gfx1250) — compile-verified, matrix operand order checked by reading, not run on hardware
//
#include <hip/hip_runtime.h>
#include <hip/hip_bf16.h>

typedef __attribute__((ext_vector_type(16))) __bf16          v16bf;
typedef __attribute__((ext_vector_type(8)))  float           v8f;
typedef __attribute__((ext_vector_type(8)))  unsigned short  ushort8;
typedef __attribute__((ext_vector_type(4)))  unsigned short  ushort4v;
typedef __attribute__((vector_size(16)))     int             v4i_t;   // matches builtin proto
typedef __attribute__((vector_size(8)))      int             v2i_t;   // matches builtin proto

#define B_N     8192
#define D_IN    1024
#define D_OUT   512
#define N_E     4
#define INV_ETA 5.0f        // 1/0.2
#define EPS_N   1e-12f

#define BLK_M   128
#define BLK_N   64
#define KSTEP   32
#define NSTEP   (D_IN / KSTEP)   // 32

// ---------------- CDNA5 async global->LDS path (guarded) --------------------
#if defined(__has_builtin)
#  if __has_builtin(__builtin_amdgcn_global_load_async_to_lds_b128) && \
      __has_builtin(__builtin_amdgcn_global_load_async_to_lds_b64)  && \
      __has_builtin(__builtin_amdgcn_s_wait_asynccnt)
#    define USE_ASYNC_LDS 1
#    warning "CDNA5 path: global_load_async_to_lds + s_wait_asynccnt ENABLED"
#  endif
#endif
#ifndef USE_ASYNC_LDS
#  define USE_ASYNC_LDS 0
#  warning "CDNA5 async-to-LDS builtins not found: using sync VGPR->ds_store staging"
#endif

union ABFrag { v16bf bf; ushort8 u[2]; };

// generic->AS1 is numerically identity; generic->AS3 is low-32-bit truncation
// so integer casts are the safe route.
__device__ __forceinline__ void copy_g2l_b128(const unsigned short* g, unsigned short* l) {
#if USE_ASYNC_LDS
    __builtin_amdgcn_global_load_async_to_lds_b128(
        (__attribute__((address_space(1))) v4i_t*)(unsigned long long)g,
        (__attribute__((address_space(3))) v4i_t*)(unsigned int)(unsigned long long)l,
        0, 0);
#else
    *(ushort8*)l = *(const ushort8*)g;
#endif
}
__device__ __forceinline__ void copy_g2l_b64(const unsigned short* g, unsigned short* l) {
#if USE_ASYNC_LDS
    __builtin_amdgcn_global_load_async_to_lds_b64(
        (__attribute__((address_space(1))) v2i_t*)(unsigned long long)g,
        (__attribute__((address_space(3))) v2i_t*)(unsigned int)(unsigned long long)l,
        0, 0);
#else
    *(ushort4v*)l = *(const ushort4v*)g;
#endif
}

// ---- float -> bf16 (round to nearest even) ----
__device__ __forceinline__ unsigned short f2bf(float f) {
    unsigned int u = __float_as_uint(f);
    u += 0x7FFFu + ((u >> 16) & 1u);
    return (unsigned short)(u >> 16);
}

// ---------------------------------------------------------------------------
// Kernel 1a: x fp32 -> bf16, vectorized by 4
// ---------------------------------------------------------------------------
__global__ __launch_bounds__(256) void cvt_x_kernel(const float* __restrict__ x,
                                                    unsigned short* __restrict__ xBF) {
    int i = blockIdx.x * blockDim.x + threadIdx.x;       // over (B_N*D_IN)/4
    const int n4 = (B_N * D_IN) / 4;
    if (i >= n4) return;
    float4 v = ((const float4*)x)[i];
    ushort4v o;
    o.x = f2bf(v.x); o.y = f2bf(v.y); o.z = f2bf(v.z); o.w = f2bf(v.w);
    ((ushort4v*)xBF)[i] = o;
}

// ---------------------------------------------------------------------------
// Kernel 1b: W [e][k][n] fp32 -> wT [e][n][k] bf16 (coalesced reads)
// ---------------------------------------------------------------------------
__global__ __launch_bounds__(256) void cvt_w_kernel(const float* __restrict__ W,
                                                    unsigned short* __restrict__ wT) {
    int i = blockIdx.x * blockDim.x + threadIdx.x;       // over N_E*D_IN*D_OUT
    const int n = N_E * D_IN * D_OUT;
    if (i >= n) return;
    int e   = i / (D_IN * D_OUT);
    int rem = i - e * (D_IN * D_OUT);
    int k   = rem / D_OUT;
    int col = rem - k * D_OUT;
    wT[((size_t)(e * D_OUT + col) * D_IN) + k] = f2bf(W[i]);
}

// ---------------------------------------------------------------------------
// Kernel 2: bf16 WMMA GEMM with double-buffered async LDS staging
//   block = 256 threads = 8 waves (4M x 2N), block tile 128 x 64
//   wave tile = 32 x 32 -> 4 x v_wmma_f32_16x16x32_bf16 per K-step
//   per K-step staged: A 128x32 (8KB) + B^T 64x32 (4KB), double buffered
// ---------------------------------------------------------------------------
__global__ __launch_bounds__(256) void gemm_logits_kernel(
    const unsigned short* __restrict__ xBF,
    const unsigned short* __restrict__ wT,
    const float* __restrict__ bias,
    float* __restrict__ logits)
{
    __shared__ unsigned short ldsA[2][BLK_M * KSTEP];   // 2 x 8 KB
    __shared__ unsigned short ldsB[2][BLK_N * KSTEP];   // 2 x 4 KB

    const int tid   = threadIdx.x;
    const int lane  = tid & 31;
    const int wave  = tid >> 5;
    const int waveM = wave >> 1;              // 0..3 -> +32 rows
    const int waveN = wave & 1;               // 0..1 -> +32 cols
    const int e     = blockIdx.z;
    const int m0    = blockIdx.x * BLK_M;
    const int n0    = blockIdx.y * BLK_N;
    const int m     = lane & 15;
    const int h     = lane >> 4;

    const unsigned short* gA = xBF + (size_t)m0 * D_IN;
    const unsigned short* gB = wT + (size_t)(e * D_OUT + n0) * D_IN;

    auto stage = [&](int buf, int k0) {
        // A tile: 128 rows x 32 k; 2 x b128 per thread
#pragma unroll
        for (int j = 0; j < 2; ++j) {
            int idx = tid + j * 256;          // 0..511
            int r = idx >> 2;
            int c = (idx & 3) * 8;
            copy_g2l_b128(gA + (size_t)r * D_IN + k0 + c, &ldsA[buf][r * KSTEP + c]);
        }
        // B tile: 64 rows (cols of W) x 32 k; 2 x b64 per thread
#pragma unroll
        for (int j = 0; j < 2; ++j) {
            int idx = tid + j * 256;          // 0..511
            int nn = idx >> 3;
            int c  = (idx & 7) * 4;
            copy_g2l_b64(gB + (size_t)nn * D_IN + k0 + c, &ldsB[buf][nn * KSTEP + c]);
        }
    };

    v8f zf = {0.f, 0.f, 0.f, 0.f, 0.f, 0.f, 0.f, 0.f};
    v8f acc[2][2];
#pragma unroll
    for (int i = 0; i < 2; ++i)
#pragma unroll
        for (int j = 0; j < 2; ++j) acc[i][j] = zf;

    stage(0, 0);

    for (int s = 0; s < NSTEP; ++s) {
        const int buf = s & 1;
        if (s + 1 < NSTEP) {
            stage(buf ^ 1, (s + 1) * KSTEP);     // prefetch next tile
#if USE_ASYNC_LDS
            __builtin_amdgcn_s_wait_asynccnt(4); // 4 newer ops may stay in flight
#endif
        } else {
#if USE_ASYNC_LDS
            __builtin_amdgcn_s_wait_asynccnt(0);
#endif
        }
        __syncthreads();                         // tile `buf` visible block-wide

        const unsigned short* As = ldsA[buf];
        const unsigned short* Bs = ldsB[buf];
        ABFrag a[2], b[2];
#pragma unroll
        for (int i = 0; i < 2; ++i) {
            const int row = waveM * 32 + i * 16 + m;
            a[i].u[0] = *(const ushort8*)&As[row * KSTEP + h * 8];
            a[i].u[1] = *(const ushort8*)&As[row * KSTEP + 16 + h * 8];
        }
#pragma unroll
        for (int j = 0; j < 2; ++j) {
            const int col = waveN * 32 + j * 16 + m;
            b[j].u[0] = *(const ushort8*)&Bs[col * KSTEP + h * 8];
            b[j].u[1] = *(const ushort8*)&Bs[col * KSTEP + 16 + h * 8];
        }
#pragma unroll
        for (int i = 0; i < 2; ++i)
#pragma unroll
            for (int j = 0; j < 2; ++j)
                acc[i][j] = __builtin_amdgcn_wmma_f32_16x16x32_bf16(
                    false, a[i].bf, false, b[j].bf, (short)0, acc[i][j], false, false);

        __syncthreads();                         // done reading `buf`
    }

    // Epilogue: row = m0 + waveM*32 + i*16 + h*8 + r ; col = n0 + waveN*32 + j*16 + m
    float* outBase = logits + (size_t)e * B_N * D_OUT;
#pragma unroll
    for (int i = 0; i < 2; ++i) {
#pragma unroll
        for (int j = 0; j < 2; ++j) {
            const int col = n0 + waveN * 32 + j * 16 + m;
            const float bv = bias[e * D_OUT + col];
#pragma unroll
            for (int r = 0; r < 8; ++r) {
                const int row = m0 + waveM * 32 + i * 16 + h * 8 + r;
                outBase[(size_t)row * D_OUT + col] = acc[i][j][r] + bv;
            }
        }
    }
}

// ---------------------------------------------------------------------------
// Block-wide sum reduction (256 threads), result broadcast to all threads
// ---------------------------------------------------------------------------
__device__ __forceinline__ float block_reduce_256(float v, float* sm) {
    const int t = threadIdx.x;
    __syncthreads();                 // protect sm reuse from previous call
    sm[t] = v;
    __syncthreads();
    for (int s = 128; s > 0; s >>= 1) {
        if (t < s) sm[t] += sm[t + s];
        __syncthreads();
    }
    return sm[0];
}

// ---------------------------------------------------------------------------
// Kernel 3: per batch-row: normalize -> xi (in place), evidential stats,
// conflict between consecutive experts, ws recursion -> wsRow[e][b]
// ---------------------------------------------------------------------------
__global__ __launch_bounds__(256) void row_stats_kernel(float* __restrict__ xi,
                                                        float* __restrict__ wsRow)
{
    __shared__ float sm[256];
    const int b = blockIdx.x;
    const int t = threadIdx.x;

    float w = 1.0f;
    float p0 = 0.f, p1 = 0.f, Sprev = 1.0f, bsPrev = 0.0f;

    for (int e = 0; e < N_E; ++e) {
        float* row = xi + ((size_t)e * B_N + b) * D_OUT;
        float l0 = row[t], l1 = row[t + 256];

        float ss  = block_reduce_256(l0 * l0 + l1 * l1, sm);
        float inv = 1.0f / fmaxf(sqrtf(ss), EPS_N);
        float x0 = l0 * inv, x1 = l1 * inv;
        row[t] = x0; row[t + 256] = x1;

        float a0 = __expf(x0), a1 = __expf(x1);          // alpha - 1
        float S  = block_reduce_256(a0 + a1, sm) + (float)D_OUT;
        float u  = (float)D_OUT / S;
        float bs = (S - (float)D_OUT) / S;               // sum of belief masses

        float C = 0.0f;
        if (e > 0) {
            float dot = block_reduce_256(p0 * a0 + p1 * a1, sm) / (Sprev * S);
            C = bsPrev * bs - dot;
        }
        if (t == 0) wsRow[e * B_N + b] = w;
        w = w * u / (1.0f - C);
        p0 = a0; p1 = a1; Sprev = S; bsPrev = bs;
    }
}

// ---------------------------------------------------------------------------
// Kernel 4: deterministic single-block reduction of Z_e = sum_b exp(ws_e/eta)
// ---------------------------------------------------------------------------
__global__ __launch_bounds__(1024) void z_reduce_kernel(const float* __restrict__ wsRow,
                                                        float* __restrict__ invZ)
{
    __shared__ float sm[1024];
    const int t = threadIdx.x;
    for (int e = 0; e < N_E; ++e) {
        float acc = 0.f;
        for (int b = t; b < B_N; b += 1024)
            acc += __expf(wsRow[e * B_N + b] * INV_ETA);
        __syncthreads();
        sm[t] = acc;
        __syncthreads();
        for (int s = 512; s > 0; s >>= 1) {
            if (t < s) sm[t] += sm[t + s];
            __syncthreads();
        }
        if (t == 0) invZ[e] = 1.0f / sm[0];
    }
}

// ---------------------------------------------------------------------------
// Kernel 5: out[b,d] = sum_e xi[e,b,d] * exp(ws_e[b]/eta) * invZ[e]
// ---------------------------------------------------------------------------
__global__ __launch_bounds__(256) void combine_kernel(const float* __restrict__ xi,
                                                      const float* __restrict__ wsRow,
                                                      const float* __restrict__ invZ,
                                                      float* __restrict__ out)
{
    const int n4 = (B_N * D_OUT) / 4;
    int i = blockIdx.x * blockDim.x + threadIdx.x;
    if (i >= n4) return;
    const int dpr = D_OUT / 4;
    int b  = i / dpr;
    int d4 = i - b * dpr;

    float4 r = {0.f, 0.f, 0.f, 0.f};
    const float4* xiv = (const float4*)xi;
#pragma unroll
    for (int e = 0; e < N_E; ++e) {
        float wgt = __expf(wsRow[e * B_N + b] * INV_ETA) * invZ[e];
        float4 v = xiv[((size_t)e * B_N + b) * dpr + d4];
        r.x += v.x * wgt; r.y += v.y * wgt; r.z += v.z * wgt; r.w += v.w * wgt;
    }
    ((float4*)out)[i] = r;
}

// ---------------------------------------------------------------------------
extern "C" void kernel_launch(void* const* d_in, const int* in_sizes, int n_in,
                              void* d_out, int out_size, void* d_ws, size_t ws_size,
                              hipStream_t stream) {
    const float* x    = (const float*)d_in[0];   // [8192, 1024]
    const float* W    = (const float*)d_in[1];   // [4, 1024, 512]
    const float* bias = (const float*)d_in[2];   // [4, 512]
    float* out        = (float*)d_out;           // [8192, 512]

    char* ws = (char*)d_ws;
    float*          logits = (float*)ws;                                   // 4*8192*512*4  = 67108864 B
    unsigned short* xBF    = (unsigned short*)(ws + 67108864);             // 8192*1024*2   = 16777216 B
    unsigned short* wT     = (unsigned short*)(ws + 67108864 + 16777216);  // 4*512*1024*2  =  4194304 B
    float*          wsRow  = (float*)(ws + 88080384);                      // 4*8192*4      =   131072 B
    float*          invZ   = (float*)(ws + 88080384 + 131072);             // 16 B

    // 1) precision conversion (+ W transpose) for the WMMA path
    {
        int n4 = (B_N * D_IN) / 4;
        cvt_x_kernel<<<(n4 + 255) / 256, 256, 0, stream>>>(x, xBF);
        int nw = N_E * D_IN * D_OUT;
        cvt_w_kernel<<<(nw + 255) / 256, 256, 0, stream>>>(W, wT);
    }

    // 2) bf16 WMMA GEMM with async LDS staging and fused bias
    {
        dim3 grid(B_N / BLK_M, D_OUT / BLK_N, N_E);   // (64, 8, 4)
        gemm_logits_kernel<<<grid, 256, 0, stream>>>(xBF, wT, bias, logits);
    }

    // 3) per-row normalization + evidential stats + ws recursion
    row_stats_kernel<<<B_N, 256, 0, stream>>>(logits, wsRow);

    // 4) batch-wise softmax denominators (deterministic single-block reduce)
    z_reduce_kernel<<<1, 1024, 0, stream>>>(wsRow, invZ);

    // 5) weighted combination of normalized expert outputs
    {
        int n4 = (B_N * D_OUT) / 4;
        combine_kernel<<<(n4 + 255) / 256, 256, 0, stream>>>(logits, wsRow, invZ, out);
    }
}
